// ScaledDotProdAttn_67138928771128
// MI455X (gfx1250) — matrix operand order repaired and checked by reading, not compiled, not run on hardware
//
#include <hip/hip_runtime.h>
#include <cstdint>
#include <cstddef>

// Scaled dot-product attention, fused one-pass over attn.
//   d_in: q[16,2048,128] f32, k[...], v[...]
//   d_out: out[16,2048,128] f32  ++  attn[16,2048,2048] f32
//   d_ws:  qh (f16, 8.39MB) | kh (f16, 8.39MB) | vt (f16 transposed [b][d][s], 8.39MB)
// Requires ws_size >= 25,165,824 bytes.

typedef __attribute__((ext_vector_type(16))) _Float16 v16h;
typedef __attribute__((ext_vector_type(8)))  float    v8f;

constexpr int kB = 16;
constexpr int kS = 2048;
constexpr int kD = 128;
constexpr int SSTR = 2052;  // f32 score row stride (pad: 4 banks/row -> conflict-free)
constexpr int PSTR = 2056;  // f16 prob  row stride (pad)
constexpr float kScale = 0.08838834764831845f;  // 1/sqrt(128)

union AFrag {
  uint32_t u[8];
  v16h h;
};

// ---------------- pre-pass: f32 -> f16 conversion ----------------

__global__ __launch_bounds__(256) void cvt_qk_kernel(const float* __restrict__ q,
                                                     const float* __restrict__ k,
                                                     _Float16* __restrict__ qh,
                                                     _Float16* __restrict__ kh,
                                                     int n4) {
  int i = blockIdx.x * blockDim.x + threadIdx.x;
  if (i >= n4) return;
  float4 fq = reinterpret_cast<const float4*>(q)[i];
  float4 fk = reinterpret_cast<const float4*>(k)[i];
  union { _Float16 h[4]; uint2 u; } pq, pk;
  pq.h[0] = (_Float16)fq.x; pq.h[1] = (_Float16)fq.y;
  pq.h[2] = (_Float16)fq.z; pq.h[3] = (_Float16)fq.w;
  pk.h[0] = (_Float16)fk.x; pk.h[1] = (_Float16)fk.y;
  pk.h[2] = (_Float16)fk.z; pk.h[3] = (_Float16)fk.w;
  reinterpret_cast<uint2*>(qh)[i] = pq.u;
  reinterpret_cast<uint2*>(kh)[i] = pk.u;
}

// v[b][s][d] f32 -> vt[b][d][s] f16 (tiled transpose through LDS)
__global__ __launch_bounds__(256) void cvt_vt_kernel(const float* __restrict__ v,
                                                     _Float16* __restrict__ vt) {
  __shared__ _Float16 tile[32][33];
  const int b  = blockIdx.z;
  const int s0 = blockIdx.x * 32;
  const int d0 = blockIdx.y * 32;
  const int tx = threadIdx.x;  // 0..31
  const int ty = threadIdx.y;  // 0..7
#pragma unroll
  for (int j = 0; j < 4; ++j) {
    int srow = ty + j * 8;
    tile[srow][tx] = (_Float16)v[((size_t)(b * kS + s0 + srow)) * kD + d0 + tx];
  }
  __syncthreads();
#pragma unroll
  for (int j = 0; j < 4; ++j) {
    int drow = ty + j * 8;
    vt[((size_t)(b * kD + d0 + drow)) * kS + s0 + tx] = tile[tx][drow];
  }
}

// ---------------- fused attention kernel ----------------
// Grid: (S/16, B). Block: 256 threads (8 waves). One 16-query-row stripe per WG.

__global__ __launch_bounds__(256) void attn_fused_kernel(const _Float16* __restrict__ qh,
                                                         const _Float16* __restrict__ kh,
                                                         const _Float16* __restrict__ vt,
                                                         float* __restrict__ outp,
                                                         float* __restrict__ attnp) {
  __shared__ float    sc[16 * SSTR];       // 131,328 B: raw scores / exp values
  __shared__ _Float16 pf[16 * PSTR];       //  65,792 B: normalized probs, f16
  __shared__ float    red[256];
  __shared__ float    rowinv[16];

  const int tid  = threadIdx.x;
  const int lane = tid & 31;
  const int wave = tid >> 5;   // 0..7
  const int mn   = lane & 15;  // row (A) / col (B,C,D) index
  const int hi   = lane >> 4;  // half-wave
  const int b    = blockIdx.y;
  const int q0   = blockIdx.x * 16;

  // ---- A fragments: Q rows q0..q0+15, 4 chunks of K(=d)=32 ----
  // 16-bit A layout: VGPR v holds K = (v<4?0:16) + hi*8 + (v&3)*2 (+1 packed)
  AFrag aq[4];
  {
    const _Float16* qrow = qh + ((size_t)(b * kS + q0 + mn)) * kD;
#pragma unroll
    for (int c = 0; c < 4; ++c) {
#pragma unroll
      for (int vv = 0; vv < 8; ++vv) {
        int ko = ((vv & 4) ? 16 : 0) + hi * 8 + (vv & 3) * 2;
        aq[c].u[vv] = *reinterpret_cast<const uint32_t*>(qrow + c * 32 + ko);
      }
    }
  }

  // ---- GEMM1: scores = Q K^T * scale  (each wave: 16 k-tiles of 16) ----
#pragma unroll 1
  for (int i = 0; i < 16; ++i) {
    const int kt = i * 8 + wave;
    const _Float16* krow = kh + ((size_t)(b * kS + kt * 16 + mn)) * kD;
    AFrag bk[4];
#pragma unroll
    for (int c = 0; c < 4; ++c) {
#pragma unroll
      for (int vv = 0; vv < 8; ++vv) {
        // 16-bit B layout: VGPR v holds K = (v<4?0:16) + hi*8 + (v&3)*2 (+1 packed)
        int ko = ((vv & 4) ? 16 : 0) + hi * 8 + (vv & 3) * 2;
        bk[c].u[vv] = *reinterpret_cast<const uint32_t*>(krow + c * 32 + ko);
      }
    }
    v8f acc = {};
#pragma unroll
    for (int c = 0; c < 4; ++c)
      acc = __builtin_amdgcn_wmma_f32_16x16x32_f16(false, aq[c].h, false, bk[c].h,
                                                   (short)0, acc, false, false);
    // C/D layout: VGPR r -> (M = r + 8*hi, N = mn)
#pragma unroll
    for (int r = 0; r < 8; ++r)
      sc[(r + hi * 8) * SSTR + kt * 16 + mn] = acc[r] * kScale;
  }
  __syncthreads();

  // ---- softmax over each of the 16 rows (16 threads per row) ----
  const int rrow = tid >> 4;
  const int j    = tid & 15;
  float lmax = -3.0e38f;
  for (int c = j * 4; c < kS; c += 64) {
    float4 s4 = *reinterpret_cast<const float4*>(&sc[rrow * SSTR + c]);
    lmax = fmaxf(lmax, fmaxf(fmaxf(s4.x, s4.y), fmaxf(s4.z, s4.w)));
  }
  red[tid] = lmax;  // same-wave producer/consumer below (LDS in-order per wave)
  float rowmax = -3.0e38f;
#pragma unroll
  for (int t = 0; t < 16; ++t) rowmax = fmaxf(rowmax, red[rrow * 16 + t]);

  float lsum = 0.0f;
  for (int c = j * 4; c < kS; c += 64) {
    float4 s4 = *reinterpret_cast<const float4*>(&sc[rrow * SSTR + c]);
    s4.x = __expf(s4.x - rowmax);
    s4.y = __expf(s4.y - rowmax);
    s4.z = __expf(s4.z - rowmax);
    s4.w = __expf(s4.w - rowmax);
    *reinterpret_cast<float4*>(&sc[rrow * SSTR + c]) = s4;
    lsum += (s4.x + s4.y) + (s4.z + s4.w);
  }
  red[tid] = lsum;
  float rowsum = 0.0f;
#pragma unroll
  for (int t = 0; t < 16; ++t) rowsum += red[rrow * 16 + t];
  if (j == 0) rowinv[rrow] = 1.0f / rowsum;
  __syncthreads();

  // ---- normalize: stream attn to HBM (coalesced f32x4), stash f16 in LDS ----
  {
    float* attn_base = attnp + ((size_t)(b * kS + q0)) * kS;
#pragma unroll 1
    for (int base = tid * 4; base < 16 * kS; base += 256 * 4) {
      int r = base >> 11;          // row (2048 cols per row)
      int c = base & (kS - 1);
      float inv = rowinv[r];
      float4 s4 = *reinterpret_cast<const float4*>(&sc[r * SSTR + c]);
      float4 p;
      p.x = s4.x * inv; p.y = s4.y * inv; p.z = s4.z * inv; p.w = s4.w * inv;
      *reinterpret_cast<float4*>(attn_base + (size_t)r * kS + c) = p;
      union { _Float16 h[4]; uint2 u; } pk;
      pk.h[0] = (_Float16)p.x; pk.h[1] = (_Float16)p.y;
      pk.h[2] = (_Float16)p.z; pk.h[3] = (_Float16)p.w;
      *reinterpret_cast<uint2*>(&pf[r * PSTR + c]) = pk.u;
    }
  }
  __syncthreads();

  // ---- GEMM2: out = P @ V (each wave: one 16-wide d-column tile, K=2048) ----
  {
    const int d0 = wave * 16;
    const _Float16* vrow = vt + ((size_t)(b * kD + d0 + mn)) * kS;  // V^T row = d column
    v8f acc = {};
#pragma unroll 1
    for (int kc = 0; kc < 64; ++kc) {
      AFrag ap, bv;
#pragma unroll
      for (int vv = 0; vv < 8; ++vv) {
        int koA = kc * 32 + ((vv & 4) ? 16 : 0) + hi * 8 + (vv & 3) * 2;
        int koB = kc * 32 + ((vv & 4) ? 16 : 0) + hi * 8 + (vv & 3) * 2;
        ap.u[vv] = *reinterpret_cast<const uint32_t*>(&pf[mn * PSTR + koA]);
        bv.u[vv] = *reinterpret_cast<const uint32_t*>(vrow + koB);
      }
      acc = __builtin_amdgcn_wmma_f32_16x16x32_f16(false, ap.h, false, bv.h,
                                                   (short)0, acc, false, false);
    }
#pragma unroll
    for (int r = 0; r < 8; ++r)
      outp[((size_t)(b * kS + q0 + r + hi * 8)) * kD + d0 + mn] = acc[r];
  }
}

// ---------------- launch ----------------

extern "C" void kernel_launch(void* const* d_in, const int* in_sizes, int n_in,
                              void* d_out, int out_size, void* d_ws, size_t ws_size,
                              hipStream_t stream) {
  (void)in_sizes; (void)n_in; (void)out_size; (void)ws_size;
  const float* q = (const float*)d_in[0];
  const float* k = (const float*)d_in[1];
  const float* v = (const float*)d_in[2];

  float* outp  = (float*)d_out;
  float* attnp = outp + (size_t)kB * kS * kD;

  const size_t nElem = (size_t)kB * kS * kD;  // 4,194,304
  _Float16* qh = (_Float16*)d_ws;
  _Float16* kh = qh + nElem;
  _Float16* vt = kh + nElem;

  // 1) convert q,k to f16 (row-major)
  int n4 = (int)(nElem / 4);
  cvt_qk_kernel<<<dim3((n4 + 255) / 256), dim3(256), 0, stream>>>(q, k, qh, kh, n4);

  // 2) convert v to f16, transposed to [b][d][s]
  cvt_vt_kernel<<<dim3(kS / 32, kD / 32, kB), dim3(32, 8), 0, stream>>>(v, vt);

  // 3) fused attention: one 16-row stripe per workgroup
  attn_fused_kernel<<<dim3(kS / 16, kB), dim3(256), 0, stream>>>(qh, kh, vt, outp, attnp);
}
